// Attention_5299989643989
// MI455X (gfx1250) — matrix operand order repaired and checked by reading, not compiled, not run on hardware
//
#include <hip/hip_runtime.h>
#include <hip/hip_bf16.h>

typedef _Float16 h8  __attribute__((ext_vector_type(8)));
typedef _Float16 h16 __attribute__((ext_vector_type(16)));
typedef float    f8  __attribute__((ext_vector_type(8)));

#define BB 2
#define TT 2048
#define CC 1024
#define NH 16
#define NKV 4
#define HD 64
#define KVC (NKV*HD)

// ---------------------------------------------------------------------------
// fp32 -> fp16 conversion (grid-stride)
// ---------------------------------------------------------------------------
__global__ void cvt_f32_f16(const float* __restrict__ s, _Float16* __restrict__ d, int n) {
    for (int i = blockIdx.x * blockDim.x + threadIdx.x; i < n; i += gridDim.x * blockDim.x)
        d[i] = (_Float16)s[i];
}

// ---------------------------------------------------------------------------
// Generic WMMA GEMM:  C[M x N] (f32) = A[M x K] (f16) * W[N x K]^T (f16)
// Block = 256 threads = 8 waves; block tile 32 (rows) x 64 (cols);
// each wave computes one 16x16 tile with v_wmma_f32_16x16x32_f16.
// ---------------------------------------------------------------------------
__global__ __launch_bounds__(256) void gemm_wmma_f16(
    const _Float16* __restrict__ A,   // M x K row-major
    const _Float16* __restrict__ W,   // N x K row-major (B = W^T)
    float* __restrict__ Cout,         // M x N row-major
    int M, int N, int K)
{
    const int lane = threadIdx.x & 31;
    const int wave = threadIdx.x >> 5;
    const int lhi  = lane >> 4;       // 0 or 1 (half-wave)
    const int llo  = lane & 15;
    const int rbase = blockIdx.y * 32 + (wave >> 2) * 16;
    const int cbase = blockIdx.x * 64 + (wave & 3) * 16;

    const _Float16* __restrict__ arow = A + (size_t)(rbase + llo) * K;
    const _Float16* __restrict__ wrow = W + (size_t)(cbase + llo) * K;

    f8 acc = {};
    for (int k0 = 0; k0 < K; k0 += 32) {
        __builtin_prefetch(arow + k0 + 128, 0, 0);  // global_prefetch_b8
        // A fragment: 16-bit A 16x32 layout: lanes 0-15 hold K 0..7 & 16..23,
        // lanes 16-31 hold K 8..15 & 24..31 (two contiguous b128 loads).
        const h8* a0 = (const h8*)(arow + k0 + (lhi ? 8 : 0));
        const h8* a1 = (const h8*)(arow + k0 + 16 + (lhi ? 8 : 0));
        // B fragment: lanes 0-15 hold K 0..7 & 16..23, lanes 16-31 K 8..15 & 24..31 for col llo.
        const h8* b0 = (const h8*)(wrow + k0 + (lhi ? 8 : 0));
        const h8* b1 = (const h8*)(wrow + k0 + 16 + (lhi ? 8 : 0));
        h16 a, b;
#pragma unroll
        for (int i = 0; i < 8; ++i) {
            a[i] = (*a0)[i]; a[8 + i] = (*a1)[i];
            b[i] = (*b0)[i]; b[8 + i] = (*b1)[i];
        }
        acc = __builtin_amdgcn_wmma_f32_16x16x32_f16(false, a, false, b,
                                                     (short)0, acc, false, false);
    }
    // C/D layout: VGPR r -> M = r + 8*lhi, N = llo
    float* crow = Cout + (size_t)(rbase + 8 * lhi) * N + cbase + llo;
#pragma unroll
    for (int r = 0; r < 8; ++r) crow[(size_t)r * N] = acc[r];
}

// ---------------------------------------------------------------------------
// Rotary + RMS-norm epilogue. One wave per (b, t, head) row.
// raw:  [B][T][nh][64] f32   ->   outh: [B][nh][T][64] f16 (head-major)
// ---------------------------------------------------------------------------
__global__ __launch_bounds__(128) void rot_rms(
    const float* __restrict__ raw,
    const float* __restrict__ cs,   // [T][32]
    const float* __restrict__ sn,   // [T][32]
    _Float16* __restrict__ outh,
    int nh)
{
    const int gid  = blockIdx.x * 4 + (threadIdx.x >> 5);
    const int lane = threadIdx.x & 31;
    const int total = BB * TT;               // rows per head-group handled below
    const int nrows = total * nh;
    if (gid >= nrows) return;
    const int hh = gid % nh;
    const int t  = (gid / nh) % TT;
    const int b  = gid / (nh * TT);

    const float* r0 = raw + (((size_t)b * TT + t) * nh + hh) * HD;
    float x1 = r0[lane], x2 = r0[lane + 32];
    float c = cs[t * 32 + lane], s = sn[t * 32 + lane];
    float a1 =  x1 * c + x2 * s;
    float a2 = -x1 * s + x2 * c;
    float ss = a1 * a1 + a2 * a2;
#pragma unroll
    for (int m = 1; m < 32; m <<= 1) ss += __shfl_xor(ss, m, 32);
    float inv = rsqrtf(ss * (1.0f / 64.0f) + 1e-6f) * 1.2f;
    _Float16* oo = outh + (((size_t)b * nh + hh) * TT + t) * HD;
    oo[lane]      = (_Float16)(a1 * inv);
    oo[lane + 32] = (_Float16)(a2 * inv);
}

// ---------------------------------------------------------------------------
// V epilogue: v = v_raw + 3*sigmoid(x[:12] @ Wg[kv]) * ve ; head-major f16 out
// ---------------------------------------------------------------------------
__global__ __launch_bounds__(128) void v_epilogue(
    const float* __restrict__ vraw,  // [B][T][4][64]
    const float* __restrict__ ve,    // [B][T][4][64]
    const float* __restrict__ x,     // [B][T][1024]
    const float* __restrict__ Wg,    // [4][12]
    _Float16* __restrict__ vh)       // [B][4][T][64]
{
    const int gid  = blockIdx.x * 4 + (threadIdx.x >> 5);
    const int lane = threadIdx.x & 31;
    if (gid >= BB * TT * NKV) return;
    const int kv = gid & 3;
    const int t  = (gid >> 2) % TT;
    const int b  = gid / (NKV * TT);

    const float* xr = x + ((size_t)b * TT + t) * CC;
    float z = 0.f;
#pragma unroll
    for (int i = 0; i < 12; ++i) z += xr[i] * Wg[kv * 12 + i];
    float gate = 3.0f / (1.0f + __expf(-z));

    const float* vr = vraw + (((size_t)b * TT + t) * NKV + kv) * HD;
    const float* er = ve   + (((size_t)b * TT + t) * NKV + kv) * HD;
    _Float16* oo = vh + (((size_t)b * NKV + kv) * TT + t) * HD;
    oo[lane]      = (_Float16)(vr[lane]      + gate * er[lane]);
    oo[lane + 32] = (_Float16)(vr[lane + 32] + gate * er[lane + 32]);
}

// ---------------------------------------------------------------------------
// Flash attention (causal, online softmax). 128 threads = 4 waves per block.
// Block handles 64 query rows of one (b, h); each wave owns 16 rows.
// QK^T and PV both on v_wmma_f32_16x16x32_f16. K staged row-major in LDS,
// V staged transposed (VT[d][key]) so B-fragments are contiguous b128 loads.
// ---------------------------------------------------------------------------
__global__ __launch_bounds__(128) void flash_attn(
    const _Float16* __restrict__ qh,  // [B][16][T][64]
    const _Float16* __restrict__ kh,  // [B][4][T][64]
    const _Float16* __restrict__ vh,  // [B][4][T][64]
    _Float16* __restrict__ yh)        // [B][T][1024]
{
    __shared__ _Float16 Kt[64 * 64];       // [key][d]
    __shared__ _Float16 Vt[64 * 64];       // [d][key] (transposed)
    __shared__ _Float16 Pt[4][16 * 64];    // per-wave P scratch [m][key]

    const int tile = blockIdx.x;     // 64-row q tile
    const int h    = blockIdx.y;
    const int b    = blockIdx.z;
    const int kvh  = h >> 2;
    const int tid  = threadIdx.x;
    const int wave = tid >> 5;
    const int lane = tid & 31;
    const int lhi  = lane >> 4;
    const int llo  = lane & 15;
    const int qt0  = tile * 64 + wave * 16;

    // Q A-fragments for d-chunks 0..31 and 32..63 (loaded once)
    h16 aq[2];
    {
        const _Float16* qb = qh + (((size_t)b * NH + h) * TT + qt0 + llo) * HD;
        const h8* p0 = (const h8*)(qb +      (lhi ? 8 : 0));
        const h8* p1 = (const h8*)(qb + 16 + (lhi ? 8 : 0));
        const h8* p2 = (const h8*)(qb + 32 + (lhi ? 8 : 0));
        const h8* p3 = (const h8*)(qb + 48 + (lhi ? 8 : 0));
#pragma unroll
        for (int i = 0; i < 8; ++i) {
            aq[0][i] = (*p0)[i]; aq[0][8 + i] = (*p1)[i];
            aq[1][i] = (*p2)[i]; aq[1][8 + i] = (*p3)[i];
        }
    }

    float m[8], l[8];
    f8 oa[4] = {{}, {}, {}, {}};
#pragma unroll
    for (int r = 0; r < 8; ++r) { m[r] = -1e30f; l[r] = 0.f; }

    const float scale = 0.125f;  // 1/sqrt(64)
    const int nj = tile + 1;     // causal: key blocks 0..tile

    for (int j = 0; j < nj; ++j) {
        const int k0 = j * 64;
        __syncthreads();
        // Cooperative stage: K row-major, V transposed. 2 threads per key row.
        {
            const int row = tid >> 1;
            const int off = (tid & 1) * 32;
            const _Float16* ks = kh + (((size_t)b * NKV + kvh) * TT + k0 + row) * HD + off;
            const _Float16* vs = vh + (((size_t)b * NKV + kvh) * TT + k0 + row) * HD + off;
            h8 k0v = *(const h8*)(ks);      h8 k1v = *(const h8*)(ks + 8);
            h8 k2v = *(const h8*)(ks + 16); h8 k3v = *(const h8*)(ks + 24);
            *(h8*)(&Kt[row * 64 + off])      = k0v;
            *(h8*)(&Kt[row * 64 + off + 8])  = k1v;
            *(h8*)(&Kt[row * 64 + off + 16]) = k2v;
            *(h8*)(&Kt[row * 64 + off + 24]) = k3v;
            h8 v0v = *(const h8*)(vs);      h8 v1v = *(const h8*)(vs + 8);
            h8 v2v = *(const h8*)(vs + 16); h8 v3v = *(const h8*)(vs + 24);
#pragma unroll
            for (int i = 0; i < 8; ++i) {
                Vt[(off + i) * 64 + row]      = v0v[i];
                Vt[(off + 8 + i) * 64 + row]  = v1v[i];
                Vt[(off + 16 + i) * 64 + row] = v2v[i];
                Vt[(off + 24 + i) * 64 + row] = v3v[i];
            }
        }
        __syncthreads();

        // S = Q @ K^T : 4 key-subtiles x 2 d-chunks
        f8 sc[4] = {{}, {}, {}, {}};
#pragma unroll
        for (int c = 0; c < 2; ++c) {
#pragma unroll
            for (int s = 0; s < 4; ++s) {
                const _Float16* kp = &Kt[(16 * s + llo) * 64 + 32 * c + (lhi ? 8 : 0)];
                const h8* b0 = (const h8*)kp;
                const h8* b1 = (const h8*)(kp + 16);
                h16 bk;
#pragma unroll
                for (int i = 0; i < 8; ++i) { bk[i] = (*b0)[i]; bk[8 + i] = (*b1)[i]; }
                sc[s] = __builtin_amdgcn_wmma_f32_16x16x32_f16(false, aq[c], false, bk,
                                                               (short)0, sc[s], false, false);
            }
        }

        // Online softmax per row (row M = r + 8*lhi, replicated over 16 lanes)
#pragma unroll
        for (int r = 0; r < 8; ++r) {
            const int qg = qt0 + r + 8 * lhi;
            float vmax = -1e30f;
#pragma unroll
            for (int s = 0; s < 4; ++s) {
                const int kg = k0 + 16 * s + llo;
                float val = sc[s][r] * scale;
                if (kg > qg) val = -1e30f;     // causal mask
                sc[s][r] = val;
                vmax = fmaxf(vmax, val);
            }
#pragma unroll
            for (int msk = 1; msk < 16; msk <<= 1)
                vmax = fmaxf(vmax, __shfl_xor(vmax, msk, 16));
            const float mn = fmaxf(m[r], vmax);
            const float alpha = __expf(m[r] - mn);
            float rs = 0.f;
#pragma unroll
            for (int s = 0; s < 4; ++s) {
                float p = __expf(sc[s][r] - mn);
                sc[s][r] = p;
                rs += p;
            }
#pragma unroll
            for (int msk = 1; msk < 16; msk <<= 1) rs += __shfl_xor(rs, msk, 16);
            l[r] = l[r] * alpha + rs;
            m[r] = mn;
#pragma unroll
            for (int s = 0; s < 4; ++s) oa[s][r] *= alpha;
            // C-layout -> A-layout transpose via per-wave LDS
            _Float16* pw = &Pt[wave][(r + 8 * lhi) * 64];
#pragma unroll
            for (int s = 0; s < 4; ++s) pw[16 * s + llo] = (_Float16)sc[s][r];
        }

        // O += P @ V : 2 key-chunks x 4 d-subtiles
#pragma unroll
        for (int c = 0; c < 2; ++c) {
            const _Float16* pr = &Pt[wave][llo * 64 + 32 * c];
            const h8* a0 = (const h8*)(pr + (lhi ? 8 : 0));
            const h8* a1 = (const h8*)(pr + 16 + (lhi ? 8 : 0));
            h16 ap;
#pragma unroll
            for (int i = 0; i < 8; ++i) { ap[i] = (*a0)[i]; ap[8 + i] = (*a1)[i]; }
#pragma unroll
            for (int s = 0; s < 4; ++s) {
                const _Float16* vp = &Vt[(16 * s + llo) * 64 + 32 * c + (lhi ? 8 : 0)];
                const h8* b0 = (const h8*)vp;
                const h8* b1 = (const h8*)(vp + 16);
                h16 bv;
#pragma unroll
                for (int i = 0; i < 8; ++i) { bv[i] = (*b0)[i]; bv[8 + i] = (*b1)[i]; }
                oa[s] = __builtin_amdgcn_wmma_f32_16x16x32_f16(false, ap, false, bv,
                                                               (short)0, oa[s], false, false);
            }
        }
    }

    // Finalize: O / l  ->  y_h [B][T][h*64+d] (f16, feeds Wo GEMM)
#pragma unroll
    for (int r = 0; r < 8; ++r) {
        const float inv = 1.0f / l[r];
        const int row = qt0 + r + 8 * lhi;
#pragma unroll
        for (int s = 0; s < 4; ++s)
            yh[((size_t)b * TT + row) * CC + h * HD + 16 * s + llo] =
                (_Float16)(oa[s][r] * inv);
    }
}

// ---------------------------------------------------------------------------
extern "C" void kernel_launch(void* const* d_in, const int* in_sizes, int n_in,
                              void* d_out, int out_size, void* d_ws, size_t ws_size,
                              hipStream_t stream) {
    const float* x  = (const float*)d_in[0];
    const float* ve = (const float*)d_in[1];
    const float* cs = (const float*)d_in[2];
    const float* sn = (const float*)d_in[3];
    const float* Wq = (const float*)d_in[4];
    const float* Wk = (const float*)d_in[5];
    const float* Wv = (const float*)d_in[6];
    const float* Wo = (const float*)d_in[7];
    const float* Wg = (const float*)d_in[8];
    float* out = (float*)d_out;

    char* ws = (char*)d_ws;
    size_t off = 0;
    auto alloc = [&](size_t bytes) -> char* {
        char* p = ws + off;
        off += (bytes + 255) & ~(size_t)255;
        return p;
    };
    const size_t BT = (size_t)BB * TT;
    _Float16* xh   = (_Float16*)alloc(BT * CC * 2);
    _Float16* Wqh  = (_Float16*)alloc((size_t)CC * CC * 2);
    _Float16* Wkh  = (_Float16*)alloc((size_t)KVC * CC * 2);
    _Float16* Wvh  = (_Float16*)alloc((size_t)KVC * CC * 2);
    _Float16* Woh  = (_Float16*)alloc((size_t)CC * CC * 2);
    float*    qraw = (float*)alloc(BT * CC * 4);
    float*    kraw = (float*)alloc(BT * KVC * 4);
    float*    vraw = (float*)alloc(BT * KVC * 4);
    _Float16* qhd  = (_Float16*)alloc(BT * CC * 2);   // [B][16][T][64]
    _Float16* khd  = (_Float16*)alloc(BT * KVC * 2);  // [B][4][T][64]
    _Float16* vhd  = (_Float16*)alloc(BT * KVC * 2);  // [B][4][T][64]
    _Float16* yh   = (_Float16*)alloc(BT * CC * 2);   // [B][T][1024]

    // 1) convert to f16
    cvt_f32_f16<<<1024, 256, 0, stream>>>(x,  xh,  (int)(BT * CC));
    cvt_f32_f16<<<1024, 256, 0, stream>>>(Wq, Wqh, CC * CC);
    cvt_f32_f16<<<256,  256, 0, stream>>>(Wk, Wkh, KVC * CC);
    cvt_f32_f16<<<256,  256, 0, stream>>>(Wv, Wvh, KVC * CC);
    cvt_f32_f16<<<1024, 256, 0, stream>>>(Wo, Woh, CC * CC);

    // 2) QKV projections (WMMA GEMMs)
    gemm_wmma_f16<<<dim3(CC / 64,  BT / 32), 256, 0, stream>>>(xh, Wqh, qraw, (int)BT, CC,  CC);
    gemm_wmma_f16<<<dim3(KVC / 64, BT / 32), 256, 0, stream>>>(xh, Wkh, kraw, (int)BT, KVC, CC);
    gemm_wmma_f16<<<dim3(KVC / 64, BT / 32), 256, 0, stream>>>(xh, Wvh, vraw, (int)BT, KVC, CC);

    // 3) epilogues: rotary + rmsnorm (q,k), gated ve add (v)
    rot_rms<<<(BB * TT * NH) / 4,  128, 0, stream>>>(qraw, cs, sn, qhd, NH);
    rot_rms<<<(BB * TT * NKV) / 4, 128, 0, stream>>>(kraw, cs, sn, khd, NKV);
    v_epilogue<<<(BB * TT * NKV) / 4, 128, 0, stream>>>(vraw, ve, x, Wg, vhd);

    // 4) causal flash attention (WMMA QK^T and PV)
    flash_attn<<<dim3(TT / 64, NH, BB), 128, 0, stream>>>(qhd, khd, vhd, yh);

    // 5) output projection (WMMA GEMM) -> f32 out
    gemm_wmma_f16<<<dim3(CC / 64, BT / 32), 256, 0, stream>>>(yh, Woh, out, (int)BT, CC, CC);
}
